// TransformerStrategyNet_79096117723623
// MI455X (gfx1250) — matrix operand order repaired and checked by reading, not compiled, not run on hardware
//
#include <hip/hip_runtime.h>
#include <hip/hip_bf16.h>
#include <math.h>

typedef _Float16 half_t;
typedef __attribute__((ext_vector_type(16))) _Float16 v16h;
typedef __attribute__((ext_vector_type(8)))  _Float16 v8h;
typedef __attribute__((ext_vector_type(8)))  float    v8f;

#define THREADS 256
#define WPB 8   // waves (sequences) per block

// ---- LDS layout ----
// Weights of the CURRENT layer, converted to f16 (offsets in halves):
#define WOFF_QKV 0        // 192x64
#define WOFF_O   12288    // 64x64
#define WOFF_1   16384    // 128x64
#define WOFF_2   24576    // 64x128
#define WBYTES   65536
// Bias/LN staging region (f32), offsets in floats:
#define BOFF_QKV 0        // 192
#define BOFF_O   192      // 64
#define BOFF_G1  256      // 64
#define BOFF_E1  320      // 64
#define BOFF_1   384      // 128
#define BOFF_2   512      // 64
#define BOFF_G2  576      // 64
#define BOFF_E2  640      // 64
#define BBYTES   2816     // 704 f32
// Per-wave scratch (byte offsets inside the wave region):
#define XOFF     0        // x:   16x64 f16  (2048B)
#define QKOFF    2048     // q,k: 16x128 f16 (4096B)  -- reused as FF hidden 16x128
#define VTOFF    6144     // vT:  64x16 f16  (2048B)
#define CTXOFF   8192     // ctx: 16x64 f16  (2048B)
#define SOFF     10240    // scores: 4 heads x 16x16 f16 (2048B)
#define WAVE_SCRATCH 12288
#define SMEM_BYTES (WBYTES + BBYTES + WPB * WAVE_SCRATCH)   // 166656 B

struct Ptrs { const void* p[32]; };

__device__ __forceinline__ v8f zf8() {
  v8f z;
#pragma unroll
  for (int i = 0; i < 8; ++i) z[i] = 0.0f;
  return z;
}

__device__ __forceinline__ v8f wmma16(v16h a, v16h b, v8f c) {
  return __builtin_amdgcn_wmma_f32_16x16x32_f16(false, a, false, b, (short)0, c,
                                                false, false);
}

// A fragment 16x32 f16: lane<16 holds K kb+0..7 / kb+16..23; lane>=16 holds +8.
__device__ __forceinline__ v16h load_A32(const half_t* base, int rs, int kb, int lane) {
  const int m = lane & 15, hs = lane >> 4;
  const half_t* p = base + m * rs + kb + hs * 8;
  v8h c0 = *(const v8h*)p;
  v8h c1 = *(const v8h*)(p + 16);
  v16h a;
#pragma unroll
  for (int i = 0; i < 8; ++i) { a[i] = c0[i]; a[8 + i] = c1[i]; }
  return a;
}

// A fragment with real K=16, upper K 16..31 zero-padded.
__device__ __forceinline__ v16h load_A16(const half_t* base, int rs, int off, int lane) {
  const int m = lane & 15, hs = lane >> 4;
  v8h c0 = *(const v8h*)(base + m * rs + off + hs * 8);
  v16h a;
#pragma unroll
  for (int i = 0; i < 8; ++i) { a[i] = c0[i]; a[8 + i] = (half_t)0.0f; }
  return a;
}

// B fragment 32x16 f16 from row-major W[n][k]: lane = column n = l&15,
// lanes 0-15 hold K kb+0..7 / kb+16..23; lanes 16-31 hold +8.
__device__ __forceinline__ v16h load_B32(const half_t* base, int rs, int n0, int kb, int lane) {
  const int n = lane & 15, hs = lane >> 4;
  const half_t* p = base + (n0 + n) * rs + kb + hs * 8;
  v8h c0 = *(const v8h*)p;
  v8h c1 = *(const v8h*)(p + 16);
  v16h b;
#pragma unroll
  for (int i = 0; i < 8; ++i) { b[i] = c0[i]; b[8 + i] = c1[i]; }
  return b;
}

// B fragment with real K=16 (upper K 16..31 zero-padded).
__device__ __forceinline__ v16h load_B16(const half_t* base, int rs, int lane) {
  v16h b;
#pragma unroll
  for (int i = 0; i < 16; ++i) b[i] = (half_t)0.0f;
  {
    const half_t* p = base + (lane & 15) * rs + (lane >> 4) * 8;
    v8h c0 = *(const v8h*)p;
#pragma unroll
    for (int i = 0; i < 8; ++i) { b[i] = c0[i]; }
  }
  return b;
}

// Store a D fragment (f32) as f16 columns into LDS matrix [16][rs].
__device__ __forceinline__ void store_D_col(v8f d, half_t* base, int rs, int n0, int lane) {
  const int n = lane & 15, mb = (lane >> 4) * 8;
#pragma unroll
  for (int v = 0; v < 8; ++v) base[(mb + v) * rs + n0 + n] = (half_t)d[v];
}

// Butterfly reductions confined to 16-lane halves (xor masks 1,2,4,8).
__device__ __forceinline__ float hsum16(float x) {
#pragma unroll
  for (int m = 1; m <= 8; m <<= 1) x += __shfl_xor(x, m, 32);
  return x;
}
__device__ __forceinline__ float hmax16(float x) {
#pragma unroll
  for (int m = 1; m <= 8; m <<= 1) x = fmaxf(x, __shfl_xor(x, m, 32));
  return x;
}

// residual-add with xb, per-row LayerNorm over 64 dims, write back to xb as f16.
__device__ void residual_ln_store(v8f* acc, half_t* xb, const float* g, const float* be,
                                  int lane) {
  const int n = lane & 15, mb = (lane >> 4) * 8;
#pragma unroll
  for (int t = 0; t < 4; ++t)
#pragma unroll
    for (int v = 0; v < 8; ++v)
      acc[t][v] += (float)xb[(mb + v) * 64 + t * 16 + n];
  float mean[8], rstd[8];
#pragma unroll
  for (int v = 0; v < 8; ++v) {
    float s = acc[0][v] + acc[1][v] + acc[2][v] + acc[3][v];
    mean[v] = hsum16(s) * (1.0f / 64.0f);
  }
#pragma unroll
  for (int v = 0; v < 8; ++v) {
    float s = 0.0f;
#pragma unroll
    for (int t = 0; t < 4; ++t) { float d = acc[t][v] - mean[v]; s += d * d; }
    rstd[v] = rsqrtf(hsum16(s) * (1.0f / 64.0f) + 1e-5f);
  }
#pragma unroll
  for (int t = 0; t < 4; ++t) {
    float gv = g[t * 16 + n], bv = be[t * 16 + n];
#pragma unroll
    for (int v = 0; v < 8; ++v) {
      float y = (acc[t][v] - mean[v]) * rstd[v] * gv + bv;
      xb[(mb + v) * 64 + t * 16 + n] = (half_t)y;
    }
  }
}

__device__ void encoder_layer(half_t* xb, half_t* qk, half_t* vT, half_t* cx, half_t* sS,
                              const half_t* Wq, const half_t* Wo,
                              const half_t* W1w, const half_t* W2w,
                              const float* bsh, int lane) {
  const int n = lane & 15, mb = (lane >> 4) * 8;
  const float* bqkv = bsh + BOFF_QKV;
  const float* bo   = bsh + BOFF_O;
  const float* g1   = bsh + BOFF_G1;
  const float* be1  = bsh + BOFF_E1;
  const float* b1   = bsh + BOFF_1;
  const float* b2   = bsh + BOFF_2;
  const float* g2   = bsh + BOFF_G2;
  const float* be2  = bsh + BOFF_E2;

  // ---- QKV projection: [16,64] @ Wqkv^T -> 12 tiles of 16 features ----
  {
    v16h ax0 = load_A32(xb, 64, 0, lane);   // invariant A fragments, hoisted
    v16h ax1 = load_A32(xb, 64, 32, lane);
    // q (feat 0..63) and k (64..127) -> qk[token][feat], branch-free loop
#pragma unroll 4
    for (int j = 0; j < 8; ++j) {
      v8f acc = wmma16(ax0, load_B32(Wq, 64, j * 16, 0, lane), zf8());
      acc     = wmma16(ax1, load_B32(Wq, 64, j * 16, 32, lane), acc);
      float bb = bqkv[j * 16 + n];
#pragma unroll
      for (int v = 0; v < 8; ++v) acc[v] += bb;
      store_D_col(acc, qk, 128, j * 16, lane);
    }
    // v (feat 128..191) -> transposed vT[feat][token], one packed b128 store
#pragma unroll
    for (int j = 0; j < 4; ++j) {
      v8f acc = wmma16(ax0, load_B32(Wq, 64, 128 + j * 16, 0, lane), zf8());
      acc     = wmma16(ax1, load_B32(Wq, 64, 128 + j * 16, 32, lane), acc);
      float bb = bqkv[128 + j * 16 + n];
      v8h pk;
#pragma unroll
      for (int v = 0; v < 8; ++v) pk[v] = (half_t)(acc[v] + bb);
      *(v8h*)(vT + (j * 16 + n) * 16 + mb) = pk;
    }
  }

  // ---- attention, phases batched across the 4 heads for WMMA pipelining ----
  {
    v8f sc[4];
#pragma unroll
    for (int h = 0; h < 4; ++h) {                           // 4 score WMMAs
      v16h aq = load_A16(qk, 128, h * 16, lane);            // q rows
      v16h bk = load_B16(qk + 64 + h * 16, 128, lane);      // k rows as B columns
      sc[h] = wmma16(aq, bk, zf8());
    }
#pragma unroll
    for (int h = 0; h < 4; ++h) {                           // softmax + f16 store
#pragma unroll
      for (int v = 0; v < 8; ++v) {
        float val = (n == 15) ? -1e30f : sc[h][v] * 0.25f;  // mask pad key, /sqrt(16)
        float mx = hmax16(val);
        float e = __expf(val - mx);
        float sum = hsum16(e);
        sS[h * 256 + (mb + v) * 16 + n] = (half_t)(e / sum);
      }
    }
#pragma unroll
    for (int h = 0; h < 4; ++h) {                           // 4 ctx WMMAs
      v16h as = load_A16(sS + h * 256, 16, 0, lane);        // attn probs
      v16h bv = load_B16(vT + h * 256, 16, lane);           // vT rows = B columns
      v8f c = wmma16(as, bv, zf8());
      store_D_col(c, cx, 64, h * 16, lane);
    }
  }

  // ---- Wo projection + residual + LN1 ----
  v8f o[4];
  {
    v16h ac0 = load_A32(cx, 64, 0, lane);
    v16h ac1 = load_A32(cx, 64, 32, lane);
#pragma unroll
    for (int t = 0; t < 4; ++t) {
      o[t] = wmma16(ac0, load_B32(Wo, 64, t * 16, 0, lane), zf8());
      o[t] = wmma16(ac1, load_B32(Wo, 64, t * 16, 32, lane), o[t]);
      float bb = bo[t * 16 + n];
#pragma unroll
      for (int v = 0; v < 8; ++v) o[t][v] += bb;
    }
  }
  residual_ln_store(o, xb, g1, be1, lane);

  // ---- FFN: relu(x@W1^T+b1)@W2^T+b2, residual, LN2 ----
  half_t* hb = qk;  // reuse q/k buffer for 16x128 hidden
  {
    v16h ax0 = load_A32(xb, 64, 0, lane);
    v16h ax1 = load_A32(xb, 64, 32, lane);
#pragma unroll 4
    for (int j = 0; j < 8; ++j) {
      v8f acc = wmma16(ax0, load_B32(W1w, 64, j * 16, 0, lane), zf8());
      acc     = wmma16(ax1, load_B32(W1w, 64, j * 16, 32, lane), acc);
      float bb = b1[j * 16 + n];
#pragma unroll
      for (int v = 0; v < 8; ++v) acc[v] = fmaxf(acc[v] + bb, 0.0f);
      store_D_col(acc, hb, 128, j * 16, lane);
    }
  }
  {
    v16h ah0 = load_A32(hb, 128, 0, lane);
    v16h ah1 = load_A32(hb, 128, 32, lane);
    v16h ah2 = load_A32(hb, 128, 64, lane);
    v16h ah3 = load_A32(hb, 128, 96, lane);
#pragma unroll
    for (int t = 0; t < 4; ++t) {
      o[t] = wmma16(ah0, load_B32(W2w, 128, t * 16, 0, lane), zf8());
      o[t] = wmma16(ah1, load_B32(W2w, 128, t * 16, 32, lane), o[t]);
      o[t] = wmma16(ah2, load_B32(W2w, 128, t * 16, 64, lane), o[t]);
      o[t] = wmma16(ah3, load_B32(W2w, 128, t * 16, 96, lane), o[t]);
      float bb = b2[t * 16 + n];
#pragma unroll
      for (int v = 0; v < 8; ++v) o[t][v] += bb;
    }
  }
  residual_ln_store(o, xb, g2, be2, lane);
}

__global__ __launch_bounds__(THREADS)
void tsnet_kernel(Ptrs pp, float* __restrict__ out, int Btot) {
  extern __shared__ char smem[];
  half_t* wsh = (half_t*)smem;
  float*  bsh = (float*)(smem + WBYTES);
  const int tid = threadIdx.x;
  const int wave = tid >> 5;
  const int lane = tid & 31;
  const int seq = blockIdx.x * WPB + wave;
  const bool active = (seq < Btot);

  char* ws = smem + WBYTES + BBYTES + wave * WAVE_SCRATCH;
  half_t* xb = (half_t*)(ws + XOFF);
  half_t* qk = (half_t*)(ws + QKOFF);
  half_t* vT = (half_t*)(ws + VTOFF);
  half_t* cx = (half_t*)(ws + CTXOFF);
  half_t* sS = (half_t*)(ws + SOFF);

  // ---- embedding gather + sinusoidal positional encoding -> xb (f16) ----
  if (active) {
    const int* priv_c  = (const int*)pp.p[0];
    const int* board_c = (const int*)pp.p[1];
    const int* hist    = (const int*)pp.p[2];
    const float* priv_emb  = (const float*)pp.p[3];
    const float* board_emb = (const float*)pp.p[4];
    const float* act_emb   = (const float*)pp.p[5];
#pragma unroll 1
    for (int m = 0; m < 16; ++m) {
      const float* erow = nullptr;
      if (m == 0)      erow = priv_emb  + 64 * priv_c[seq];
      else if (m == 1) erow = board_emb + 64 * board_c[seq];
      else if (m < 15) erow = act_emb   + 64 * hist[seq * 13 + (m - 2)];
#pragma unroll
      for (int c = 0; c < 2; ++c) {
        int d = lane + 32 * c;
        float val = 0.0f;
        if (erow) {
          // div = exp(-(2i)*ln(10000)/64), i = d/2
          float ang = (float)m * __expf(-0.14391157f * (float)(d & ~1));
          val = erow[d] + ((d & 1) ? __cosf(ang) : __sinf(ang));
        }
        xb[m * 64 + d] = (half_t)val;   // row 15 = zero pad
      }
    }
  }

  for (int L = 0; L < 2; ++L) {
    // cooperative f32 -> f16 weight staging into LDS (broadcast across 8 seqs)
    const float* wq = (const float*)pp.p[6 + 12 * L];
    const float* wo = (const float*)pp.p[8 + 12 * L];
    const float* w1 = (const float*)pp.p[12 + 12 * L];
    const float* w2 = (const float*)pp.p[14 + 12 * L];
    for (int i = tid; i < 12288; i += THREADS) wsh[WOFF_QKV + i] = (half_t)wq[i];
    for (int i = tid; i < 4096;  i += THREADS) wsh[WOFF_O   + i] = (half_t)wo[i];
    for (int i = tid; i < 8192;  i += THREADS) wsh[WOFF_1   + i] = (half_t)w1[i];
    for (int i = tid; i < 8192;  i += THREADS) wsh[WOFF_2   + i] = (half_t)w2[i];
    // stage biases + LN params (f32)
    const float* pbq = (const float*)pp.p[7 + 12 * L];
    const float* pbo = (const float*)pp.p[9 + 12 * L];
    const float* pg1 = (const float*)pp.p[10 + 12 * L];
    const float* pe1 = (const float*)pp.p[11 + 12 * L];
    const float* pb1 = (const float*)pp.p[13 + 12 * L];
    const float* pb2 = (const float*)pp.p[15 + 12 * L];
    const float* pg2 = (const float*)pp.p[16 + 12 * L];
    const float* pe2 = (const float*)pp.p[17 + 12 * L];
    for (int i = tid; i < 192; i += THREADS) bsh[BOFF_QKV + i] = pbq[i];
    for (int i = tid; i < 128; i += THREADS) bsh[BOFF_1   + i] = pb1[i];
    if (tid < 64) {
      bsh[BOFF_O  + tid] = pbo[tid];
      bsh[BOFF_G1 + tid] = pg1[tid];
      bsh[BOFF_E1 + tid] = pe1[tid];
      bsh[BOFF_2  + tid] = pb2[tid];
      bsh[BOFF_G2 + tid] = pg2[tid];
      bsh[BOFF_E2 + tid] = pe2[tid];
    }
    __syncthreads();
    if (active) {
      encoder_layer(xb, qk, vT, cx, sS,
                    wsh + WOFF_QKV, wsh + WOFF_O, wsh + WOFF_1, wsh + WOFF_2,
                    bsh, lane);
    }
    __syncthreads();
  }

  // ---- logits = x[token0] @ fc_w^T + fc_b ----
  if (active) {
    const float* fcw = (const float*)pp.p[30];
    const float* fcb = (const float*)pp.p[31];
    float x0 = (float)xb[lane];
    float x1 = (float)xb[lane + 32];
#pragma unroll
    for (int a = 0; a < 3; ++a) {
      float s = x0 * fcw[a * 64 + lane] + x1 * fcw[a * 64 + lane + 32];
#pragma unroll
      for (int m = 16; m >= 1; m >>= 1) s += __shfl_xor(s, m, 32);
      if (lane == 0) out[seq * 3 + a] = s + fcb[a];
    }
  }
}

extern "C" void kernel_launch(void* const* d_in, const int* in_sizes, int n_in,
                              void* d_out, int out_size, void* d_ws, size_t ws_size,
                              hipStream_t stream) {
  (void)d_ws; (void)ws_size; (void)out_size;
  Ptrs pp;
  for (int i = 0; i < 32; ++i) pp.p[i] = (i < n_in) ? d_in[i] : nullptr;
  const int B = in_sizes[0];
  const int blocks = (B + WPB - 1) / WPB;
  hipFuncSetAttribute(reinterpret_cast<const void*>(tsnet_kernel),
                      hipFuncAttributeMaxDynamicSharedMemorySize, SMEM_BYTES);
  tsnet_kernel<<<blocks, THREADS, SMEM_BYTES, stream>>>(pp, (float*)d_out, B);
}
